// AdditiveCouplingLayer_89464168775890
// MI455X (gfx1250) — matrix operand order repaired and checked by reading, not compiled, not run on hardware
//
#include <hip/hip_runtime.h>
#include <hip/hip_bf16.h>

// ---------------------------------------------------------------------------
// Additive coupling layer, fused 3-layer MLP on MI455X (gfx1250).
// bf16 WMMA (v_wmma_f32_16x16x32_bf16), f32 accumulate, bias folded into C init.
// Per-WG: 64 batch rows; activations resident in LDS (256 KB, H2 aliases As).
// Per-wave: 4 M-blocks x 4 N-tiles per pass => 16 independent WMMAs per k-step.
// ---------------------------------------------------------------------------

typedef __bf16 bf16_t;
typedef __attribute__((ext_vector_type(16))) __bf16 v16bf;
typedef __attribute__((ext_vector_type(8)))  __bf16 v8bf;
typedef __attribute__((ext_vector_type(2)))  __bf16 v2bf;
typedef __attribute__((ext_vector_type(8)))  float  v8f;

#define M_TILE 64   // batch rows per workgroup
#define BLOCK  256  // 8 waves of 32

// ---- f32 -> bf16 via native fptrunc (RNE; compiler picks HW cvt if present)
__device__ __forceinline__ bf16_t f2bf(float f) { return (bf16_t)f; }

__device__ __forceinline__ unsigned packpair(float lo, float hi) {
    v2bf pk;
    pk[0] = (bf16_t)lo;
    pk[1] = (bf16_t)hi;
    unsigned u; __builtin_memcpy(&u, &pk, 4);
    return u;
}
__device__ __forceinline__ void store_bf16_pair(bf16_t* p0, bf16_t* p1,
                                                float v0, float v1) {
    *p0 = (bf16_t)v0;
    *p1 = (bf16_t)v1;
}

// ---- single-op ReLU via v_med3_f32 when available -------------------------
#if __has_builtin(__builtin_amdgcn_fmed3f)
__device__ __forceinline__ float relu1(float v) {
    return __builtin_amdgcn_fmed3f(v, 0.0f, __builtin_inff());
}
#else
__device__ __forceinline__ float relu1(float v) { return fmaxf(v, 0.0f); }
#endif

// A-fragment (16x32 bf16, row-major in LDS), contiguous k order (k = 16h + s, same as B):
// lanes 0-15 : row=lane,     K = [k0..k0+15]
// lanes 16-31: row=lane-16,  K = [k0+16..k0+31]
__device__ __forceinline__ v16bf load_afrag(const bf16_t* __restrict__ in,
                                            int stride, int m0, int k0, int lane) {
    const int row  = m0 + (lane & 15);
    const int kadj = (lane >> 4) << 4;          // 0 or 16
    const bf16_t* p = in + (size_t)row * stride + k0 + kadj;
    v8bf lo = *(const v8bf*)(p);
    v8bf hi = *(const v8bf*)(p + 8);
    v16bf a;
#pragma unroll
    for (int i = 0; i < 8; ++i) { a[i] = lo[i]; a[i + 8] = hi[i]; }
    return a;
}

// B-fragment (32x16 bf16). Weights pre-transposed to N x K row-major =>
// one contiguous 32-byte load per lane:
// lanes 0-15: col=lane, K=k0..k0+15 ; lanes 16-31: col=lane-16, K=k0+16..k0+31
__device__ __forceinline__ v16bf load_bfrag(const bf16_t* __restrict__ wt,
                                            int Kdim, int n, int k0, int lane) {
    const int col  = n + (lane & 15);
    const int kadj = (lane >> 4) << 4;          // 0 or 16
    return *(const v16bf*)(wt + (size_t)col * Kdim + k0 + kadj);
}

// Hidden layer: LDS act (64 x KDIM, lo/hi 32-row regions) @ WT^T + bias, ReLU,
// bf16 result into LDS (lo/hi 32-row regions, stride NOUT).
template <int KDIM, int NOUT, bool RELU>
__device__ __forceinline__ void gemm_lds(const bf16_t* __restrict__ in_lo,
                                         const bf16_t* __restrict__ in_hi,
                                         const bf16_t* __restrict__ wt,
                                         const float* __restrict__ bias,
                                         bf16_t* __restrict__ out_lo,
                                         bf16_t* __restrict__ out_hi,
                                         int wave, int lane) {
    const int nspan   = NOUT / 8;               // distinct strip per wave
    const int n_start = wave * nspan;
    const int row_hi  = (lane >> 4) << 3;       // 0 or 8 (C layout)
    const int ncol    = lane & 15;

    for (int nt = 0; nt < nspan; nt += 64) {    // four 16-wide N tiles per pass
        const int nb = n_start + nt;
        v8f c[4][4];
#pragma unroll
        for (int t = 0; t < 4; ++t) {           // bias folded into C init
            const float bv = bias[nb + 16 * t + ncol];
#pragma unroll
            for (int mb = 0; mb < 4; ++mb)
#pragma unroll
                for (int j = 0; j < 8; ++j) c[mb][t][j] = bv;
        }

        for (int k0 = 0; k0 < KDIM; k0 += 32) {
            v16bf b[4];
#pragma unroll
            for (int t = 0; t < 4; ++t)
                b[t] = load_bfrag(wt, KDIM, nb + 16 * t, k0, lane);
#pragma unroll
            for (int mb = 0; mb < 4; ++mb) {
                const bf16_t* inb = (mb < 2) ? in_lo : in_hi;   // compile-time
                const v16bf a = load_afrag(inb, KDIM, (mb & 1) * 16, k0, lane);
#pragma unroll
                for (int t = 0; t < 4; ++t)
                    c[mb][t] = __builtin_amdgcn_wmma_f32_16x16x32_bf16(
                        false, a, false, b[t], (short)0, c[mb][t], false, false);
            }
        }
#pragma unroll
        for (int mb = 0; mb < 4; ++mb) {
            bf16_t* outb = (mb < 2) ? out_lo : out_hi;          // compile-time
            const int rb = (mb & 1) * 16 + row_hi;
#pragma unroll
            for (int t = 0; t < 4; ++t) {
                const int n = nb + 16 * t + ncol;
#pragma unroll
                for (int j = 0; j < 8; j += 2) {
                    float v0 = c[mb][t][j];
                    float v1 = c[mb][t][j + 1];
                    if (RELU) { v0 = relu1(v0); v1 = relu1(v1); }
                    store_bf16_pair(outb + (size_t)(rb + j) * NOUT + n,
                                    outb + (size_t)(rb + j + 1) * NOUT + n,
                                    v0, v1);
                }
            }
        }
    }
}

// Final layer (NOUT=512, single N-pass per wave) with fused coupling epilogue:
// y[:,1::2] = x[:,1::2] + (h @ W3 + b3) ; y[:,0::2] = x[:,0::2]  (float2 ops)
template <int KDIM>
__device__ __forceinline__ void gemm_final(const bf16_t* __restrict__ in_lo,
                                           const bf16_t* __restrict__ in_hi,
                                           const bf16_t* __restrict__ wt,
                                           const float* __restrict__ bias,
                                           const float* __restrict__ x,
                                           float* __restrict__ y,
                                           int row0, int wave, int lane) {
    const int NOUT    = 512;
    const int nspan   = NOUT / 8;               // 64 => one pass of 4 tiles
    const int n_start = wave * nspan;
    const int row_hi  = (lane >> 4) << 3;
    const int ncol    = lane & 15;

    for (int nt = 0; nt < nspan; nt += 64) {
        const int nb = n_start + nt;
        v8f c[4][4];
#pragma unroll
        for (int t = 0; t < 4; ++t) {
            const float bv = bias[nb + 16 * t + ncol];
#pragma unroll
            for (int mb = 0; mb < 4; ++mb)
#pragma unroll
                for (int j = 0; j < 8; ++j) c[mb][t][j] = bv;
        }

        for (int k0 = 0; k0 < KDIM; k0 += 32) {
            v16bf b[4];
#pragma unroll
            for (int t = 0; t < 4; ++t)
                b[t] = load_bfrag(wt, KDIM, nb + 16 * t, k0, lane);
#pragma unroll
            for (int mb = 0; mb < 4; ++mb) {
                const bf16_t* inb = (mb < 2) ? in_lo : in_hi;
                const v16bf a = load_afrag(inb, KDIM, (mb & 1) * 16, k0, lane);
#pragma unroll
                for (int t = 0; t < 4; ++t)
                    c[mb][t] = __builtin_amdgcn_wmma_f32_16x16x32_bf16(
                        false, a, false, b[t], (short)0, c[mb][t], false, false);
            }
        }
#pragma unroll
        for (int mb = 0; mb < 4; ++mb) {
#pragma unroll
            for (int t = 0; t < 4; ++t) {
                const int n = nb + 16 * t + ncol;
#pragma unroll
                for (int j = 0; j < 8; ++j) {
                    const int row = row0 + mb * 16 + row_hi + j;
                    const size_t base = (size_t)row * 1024 + 2 * n;
                    float2 xv = *(const float2*)(x + base);
                    float2 yv; yv.x = xv.x; yv.y = xv.y + c[mb][t][j];
                    *(float2*)(y + base) = yv;
                }
            }
        }
    }
}

// Convert W (K x N, f32) -> WT (N x K, bf16)
__global__ void __launch_bounds__(BLOCK)
wconv_kernel(const float* __restrict__ W, bf16_t* __restrict__ WT, int K, int N) {
    const int idx = blockIdx.x * BLOCK + threadIdx.x;
    if (idx >= K * N) return;
    const int n = idx / K;
    const int k = idx % K;
    WT[idx] = f2bf(W[(size_t)k * N + n]);
}

__global__ void __launch_bounds__(BLOCK, 1)
coupling_kernel(const float*  __restrict__ x,
                const bf16_t* __restrict__ wt1, const float* __restrict__ b1,
                const bf16_t* __restrict__ wt2, const float* __restrict__ b2,
                const bf16_t* __restrict__ wt3, const float* __restrict__ b3,
                float* __restrict__ y) {
    extern __shared__ char smem[];
    bf16_t* As = (bf16_t*)smem;                 // 64 x 512 (64 KB); later H2 rows 0-31
    bf16_t* H1 = As + M_TILE * 512;             // 64 x 1024 (128 KB)
    bf16_t* Hx = H1 + M_TILE * 1024;            // 32 x 1024 (64 KB); H2 rows 32-63

    const int tid  = threadIdx.x;
    const int wave = tid >> 5;
    const int lane = tid & 31;
    const int row0 = blockIdx.x * M_TILE;

    // Stage masked = x[:, 0::2] into LDS as bf16 (float4 read, packed b32 store)
    unsigned* As32 = (unsigned*)As;
    for (int i = tid; i < M_TILE * 256; i += BLOCK) {
        const int m  = i >> 8;                  // row in tile
        const int kk = i & 255;                 // pair of masked columns
        const float4 v = *(const float4*)(x + (size_t)(row0 + m) * 1024 + 4 * kk);
        As32[(m << 8) + kk] = packpair(v.x, v.z); // masked cols 2kk, 2kk+1
    }
    __syncthreads();

    gemm_lds<512, 1024, true>(As, As + 32 * 512, wt1, b1,
                              H1, H1 + 32 * 1024, wave, lane);
    __syncthreads();
    gemm_lds<1024, 1024, true>(H1, H1 + 32 * 1024, wt2, b2,
                               As, Hx, wave, lane);   // H2 aliases dead As + Hx
    __syncthreads();
    gemm_final<1024>(As, Hx, wt3, b3, x, y, row0, wave, lane);
}

extern "C" void kernel_launch(void* const* d_in, const int* in_sizes, int n_in,
                              void* d_out, int out_size, void* d_ws, size_t ws_size,
                              hipStream_t stream) {
    (void)in_sizes; (void)n_in; (void)out_size; (void)ws_size;
    const float* x  = (const float*)d_in[0];
    const float* W1 = (const float*)d_in[1];  // 512 x 1024
    const float* b1 = (const float*)d_in[2];
    const float* W2 = (const float*)d_in[3];  // 1024 x 1024
    const float* b2 = (const float*)d_in[4];
    const float* W3 = (const float*)d_in[5];  // 1024 x 512
    const float* b3 = (const float*)d_in[6];
    float* y = (float*)d_out;

    bf16_t* wt1 = (bf16_t*)d_ws;              // 1024 x 512  (N x K)
    bf16_t* wt2 = wt1 + 1024 * 512;           // 1024 x 1024
    bf16_t* wt3 = wt2 + 1024 * 1024;          // 512 x 1024

    wconv_kernel<<<(512 * 1024 + BLOCK - 1) / BLOCK, BLOCK, 0, stream>>>(W1, wt1, 512, 1024);
    wconv_kernel<<<(1024 * 1024 + BLOCK - 1) / BLOCK, BLOCK, 0, stream>>>(W2, wt2, 1024, 1024);
    wconv_kernel<<<(1024 * 512 + BLOCK - 1) / BLOCK, BLOCK, 0, stream>>>(W3, wt3, 1024, 512);

    const size_t lds_bytes =
        (size_t)(M_TILE * 512 + M_TILE * 1024 + 32 * 1024) * sizeof(bf16_t); // 256 KB
    coupling_kernel<<<65536 / M_TILE, BLOCK, lds_bytes, stream>>>(
        x, wt1, b1, wt2, b2, wt3, b3, y);
}
